// MultiHeadAttention_37177236914180
// MI455X (gfx1250) — matrix operand order repaired and checked by reading, not compiled, not run on hardware
//
#include <hip/hip_runtime.h>
#include <hip/hip_bf16.h>

// ---------------------------------------------------------------------------
// MHA forward for MI455X (gfx1250, wave32, WMMA bf16 16x16x32, f32 accum).
// Pipeline: cvt x->bf16 (vectorized), LDS-tiled transpose-cvt weights,
//           QKV GEMM (WMMA, 64x64/wave), LDS-tiled V^T build,
//           flash-attention (WMMA, transposed-S trick, no LDS, no barriers),
//           output GEMM (WMMA) -> f32.
// ---------------------------------------------------------------------------

typedef __attribute__((ext_vector_type(16))) __bf16 v16bf;
typedef __attribute__((ext_vector_type(8)))  float  v8f;
typedef __attribute__((ext_vector_type(4)))  unsigned int u32x4;
typedef __attribute__((ext_vector_type(4)))  float  f32x4;

union FragU {
  u32x4 q[2];
  v16bf v;
  unsigned short s[16];
};

static __device__ __forceinline__ unsigned short f32_bf16(float x) {
  union { float f; unsigned u; } t;
  t.f = x;
  unsigned r = t.u + 0x7FFFu + ((t.u >> 16) & 1u);  // round-to-nearest-even
  return (unsigned short)(r >> 16);
}

// A-fragment (16x32 bf16, MxK): lane m = lane&15, k = hi*16 + e
// (contiguous, matches B). Two contiguous b128 loads per lane.
static __device__ __forceinline__ v16bf load_frag_a(const unsigned short* p, int ld) {
  const int lane = threadIdx.x & 31;
  const int m = lane & 15, hi = lane >> 4;
  const unsigned short* r = p + (size_t)m * ld + hi * 16;
  FragU f;
  f.q[0] = *(const u32x4*)(r);
  f.q[1] = *(const u32x4*)(r + 8);
  return f.v;
}

// B-fragment (32x16 bf16, KxN) loaded from Bt storage (N rows x K cols,
// row-major): lane n = lane&15, k = hi*16 + e -> 32 contiguous bytes.
static __device__ __forceinline__ v16bf load_frag_b(const unsigned short* p, int ld) {
  const int lane = threadIdx.x & 31;
  const int n = lane & 15, hi = lane >> 4;
  const unsigned short* r = p + (size_t)n * ld + hi * 16;
  FragU f;
  f.q[0] = *(const u32x4*)(r);
  f.q[1] = *(const u32x4*)(r + 8);
  return f.v;
}

static __device__ __forceinline__ v8f wmma_bf16(v16bf a, v16bf b, v8f c) {
  return __builtin_amdgcn_wmma_f32_16x16x32_bf16(false, a, false, b, (short)0, c,
                                                 false, false);
}

// Exchange with lane^16 (crosses the two 16-lane halves of a wave32).
// ds_swizzle group-of-32: offset = {xor=0x10 <<10, or=0 <<5, and=0x1f}.
static __device__ __forceinline__ float swapx16(float x) {
  return __int_as_float(__builtin_amdgcn_ds_swizzle(__float_as_int(x), 0x401f));
}

// ---------------------------------------------------------------------------
// x (f32) -> bf16, 8 elements per thread: 2x b128 load, 1x b128 store.
// ---------------------------------------------------------------------------
__global__ void cvt_bf16_kernel(const float* __restrict__ src,
                                unsigned short* __restrict__ dst, int n8) {
  int i = blockIdx.x * blockDim.x + threadIdx.x;
  if (i >= n8) return;
  f32x4 a = ((const f32x4*)src)[(size_t)i * 2];
  f32x4 b = ((const f32x4*)src)[(size_t)i * 2 + 1];
  FragU pk;
#pragma unroll
  for (int r = 0; r < 4; r++) {
    pk.s[r] = f32_bf16(a[r]);
    pk.s[4 + r] = f32_bf16(b[r]);
  }
  ((u32x4*)dst)[i] = pk.q[0];
}

// ---------------------------------------------------------------------------
// src (K,N) f32 row-major -> dst (N,K) bf16 row-major, LDS 32x32 tiles so both
// global reads and writes are coalesced. grid = (N/32, K/32), block 256.
// ---------------------------------------------------------------------------
__global__ void cvt_transpose_kernel(const float* __restrict__ src,
                                     unsigned short* __restrict__ dst,
                                     int K, int N) {
  __shared__ float tile[32][33];
  const int n0 = blockIdx.x * 32, k0 = blockIdx.y * 32;
  const int tx = threadIdx.x & 31, ty = threadIdx.x >> 5;  // 8 rows per pass
#pragma unroll
  for (int r = 0; r < 32; r += 8)
    tile[ty + r][tx] = src[(size_t)(k0 + ty + r) * N + n0 + tx];
  __syncthreads();
#pragma unroll
  for (int r = 0; r < 32; r += 8)
    dst[(size_t)(n0 + ty + r) * K + k0 + tx] = f32_bf16(tile[tx][ty + r]);
}

// ---------------------------------------------------------------------------
// Vt[bh][d][t] = qkv[b*2048+t][2048 + h*64 + d], LDS-tiled (64t x 64d):
// b128 reads along d, b128 writes along t. grid = (64 bh, 32 t-tiles), 256 thr.
// ---------------------------------------------------------------------------
__global__ void build_vt_kernel(const unsigned short* __restrict__ qkv,
                                unsigned short* __restrict__ vt) {
  __shared__ unsigned short tile[64][72];  // [t_local][d], +8 pad (row = 144B)
  const int bh = blockIdx.x;
  const int b = bh >> 4, h = bh & 15;
  const int t0 = blockIdx.y * 64;
  const unsigned short* src =
      qkv + (size_t)(b * 2048 + t0) * 3072 + 2048 + h * 64;
#pragma unroll
  for (int j = 0; j < 2; ++j) {
    int idx = j * 256 + threadIdx.x;       // 512 x 8-element chunks
    int tl = idx >> 3, c8 = (idx & 7) * 8;
    *(u32x4*)(&tile[tl][c8]) = *(const u32x4*)(src + (size_t)tl * 3072 + c8);
  }
  __syncthreads();
  unsigned short* dstb = vt + (size_t)bh * 64 * 2048 + t0;
#pragma unroll
  for (int j = 0; j < 2; ++j) {
    int idx = j * 256 + threadIdx.x;
    int d = idx >> 3, t8 = (idx & 7) * 8;
    FragU pk;
#pragma unroll
    for (int r = 0; r < 8; r++) pk.s[r] = tile[t8 + r][d];
    *(u32x4*)(dstb + (size_t)d * 2048 + t8) = pk.q[0];
  }
}

// ---------------------------------------------------------------------------
// GEMM: C(M,N) = A(M,K) @ Bt(N,K)^T, bf16 in, f32 accumulate.
// Block = 128 threads = 4 independent waves; wave tile 64x64 (16 WMMA / kstep).
// ---------------------------------------------------------------------------
template <bool OUT_BF16>
__global__ void gemm_wmma(const unsigned short* __restrict__ A,
                          const unsigned short* __restrict__ Bt,
                          void* __restrict__ Cout, int M, int N, int K) {
  const int wave = threadIdx.x >> 5;
  const int lane = threadIdx.x & 31;
  const int m0 = blockIdx.x * 128 + (wave >> 1) * 64;
  const int n0 = blockIdx.y * 128 + (wave & 1) * 64;

  v8f acc[4][4];
#pragma unroll
  for (int i = 0; i < 4; i++)
#pragma unroll
    for (int j = 0; j < 4; j++)
      acc[i][j] = {0.f, 0.f, 0.f, 0.f, 0.f, 0.f, 0.f, 0.f};

  for (int k0 = 0; k0 < K; k0 += 32) {
    v16bf af[4];
#pragma unroll
    for (int i = 0; i < 4; i++)
      af[i] = load_frag_a(A + (size_t)(m0 + i * 16) * K + k0, K);
#pragma unroll
    for (int j = 0; j < 4; j++) {
      v16bf bfj = load_frag_b(Bt + (size_t)(n0 + j * 16) * K + k0, K);
#pragma unroll
      for (int i = 0; i < 4; i++)
        acc[i][j] = wmma_bf16(af[i], bfj, acc[i][j]);
    }
  }

  // Epilogue: C-tile layout lane n = lane&15, m = r + 8*(lane>>4)
  const int cn = lane & 15, chi = lane >> 4;
#pragma unroll
  for (int i = 0; i < 4; i++)
#pragma unroll
    for (int j = 0; j < 4; j++) {
#pragma unroll
      for (int r = 0; r < 8; r++) {
        const size_t idx =
            (size_t)(m0 + i * 16 + chi * 8 + r) * N + (size_t)(n0 + j * 16 + cn);
        if (OUT_BF16)
          ((unsigned short*)Cout)[idx] = f32_bf16(acc[i][j][r]);
        else
          ((float*)Cout)[idx] = acc[i][j][r];
      }
    }
}

// ---------------------------------------------------------------------------
// Flash attention, one wave = 16 queries of one (b,h). No LDS, no barriers.
// S^T = K @ Q^T  (softmax reductions become intra-lane + one swizzle)
// O^T = V^T @ P^T (P^T B-fragment built in-register via xor-16 swizzle)
// qkv: (8192, 3072) bf16 [Q|K|V], vt: (64 bh, 64 d, 2048 t) bf16
// attnout: (8192, 1024) bf16, column h*64+d
// ---------------------------------------------------------------------------
__global__ void attn_wmma(const unsigned short* __restrict__ qkv,
                          const unsigned short* __restrict__ vt,
                          unsigned short* __restrict__ attnout) {
  const int wave = threadIdx.x >> 5;
  const int lane = threadIdx.x & 31;
  const int lo = lane & 15, hi = lane >> 4;
  const int bh = blockIdx.x;
  const int b = bh >> 4, h = bh & 15;
  const int q0 = (blockIdx.y * 4 + wave) * 16;  // first query row of this wave

  const unsigned short* qbase = qkv + (size_t)b * 2048 * 3072 + h * 64;
  const unsigned short* kbase = qbase + 1024;
  const unsigned short* vtb = vt + (size_t)bh * 64 * 2048;

  // Persistent Q^T B-fragments (K-dim = d = 64 -> two frags)
  v16bf qf0 = load_frag_b(qbase + (size_t)q0 * 3072, 3072);
  v16bf qf1 = load_frag_b(qbase + (size_t)q0 * 3072 + 32, 3072);

  v8f o[4];  // O^T accum: row = d = r+8*hi (per 16-tile), col = query = lo
#pragma unroll
  for (int dt = 0; dt < 4; dt++) o[dt] = {0.f, 0.f, 0.f, 0.f, 0.f, 0.f, 0.f, 0.f};

  float runm = -1e30f, runl = 0.0f;
  const int kbmax = q0 >> 6;  // last (diagonal) 64-key block

  for (int kb = 0; kb <= kbmax; ++kb) {
    const bool masked = (kb == kbmax);
    const unsigned short* kblk = kbase + (size_t)(kb * 64) * 3072;

    // ---- All 8 S-WMMAs issued back-to-back (hazard nops overlap issue) ----
    v8f st[4];  // S^T tiles: row = key (r+8*hi within 16-tile), col = query lo
#pragma unroll
    for (int kt = 0; kt < 4; ++kt) {
      v16bf a0 = load_frag_a(kblk + (size_t)(kt * 16) * 3072, 3072);
      v16bf a1 = load_frag_a(kblk + (size_t)(kt * 16) * 3072 + 32, 3072);
      v8f z = {0.f, 0.f, 0.f, 0.f, 0.f, 0.f, 0.f, 0.f};
      z = wmma_bf16(a0, qf0, z);
      st[kt] = wmma_bf16(a1, qf1, z);
    }

    // ---- Hoist V^T A-fragments: loads overlap the softmax VALU below ----
    v16bf va[2][4];
#pragma unroll
    for (int f = 0; f < 2; ++f)
#pragma unroll
      for (int dt = 0; dt < 4; ++dt)
        va[f][dt] =
            load_frag_a(vtb + (size_t)(dt * 16) * 2048 + kb * 64 + f * 32, 2048);

    // ---- scale, causal mask, block max ----
    float s[4][8];
    float bm = -1e30f;
#pragma unroll
    for (int kt = 0; kt < 4; ++kt)
#pragma unroll
      for (int r = 0; r < 8; r++) {
        float v = st[kt][r] * 0.125f;  // 1/sqrt(64)
        if (masked) {
          int key = kb * 64 + kt * 16 + hi * 8 + r;
          v = (key > q0 + lo) ? -1e30f : v;  // select, not branch (EXEC full)
        }
        s[kt][r] = v;
        bm = fmaxf(bm, v);
      }
    // combine the two 16-lane halves (each holds half the keys per query)
    bm = fmaxf(bm, swapx16(bm));
    float newm = fmaxf(runm, bm);
    float alpha = __expf(runm - newm);

    float bs = 0.0f;
#pragma unroll
    for (int kt = 0; kt < 4; kt++)
#pragma unroll
      for (int r = 0; r < 8; r++) {
        float p = __expf(s[kt][r] - newm);
        s[kt][r] = p;
        bs += p;
      }
    bs += swapx16(bs);
    runl = runl * alpha + bs;
    runm = newm;

#pragma unroll
    for (int dt = 0; dt < 4; dt++)
#pragma unroll
      for (int r = 0; r < 8; r++) o[dt][r] *= alpha;

    // ---- O^T += V^T(64 x 32keys) @ P^T(32keys x 16q), two 32-key steps ----
#pragma unroll
    for (int f = 0; f < 2; ++f) {
      // Build P^T B-fragment from C-layout tiles T0=s[2f], T1=s[2f+1]:
      // element e: k = hi*16+e = row e of tile T_hi, col lo.
      FragU pb;
#pragma unroll
      for (int r = 0; r < 8; r++) {
        float p0 = s[2 * f][r], p1 = s[2 * f + 1][r];
        float px0 = swapx16(p0), px1 = swapx16(p1);
        pb.s[r]     = f32_bf16(hi ? px1 : p0);  // rows 0..7 of T_hi
        pb.s[r + 8] = f32_bf16(hi ? p1 : px0);  // rows 8..15 of T_hi
      }
#pragma unroll
      for (int dt = 0; dt < 4; dt++)
        o[dt] = wmma_bf16(va[f][dt], pb.v, o[dt]);
    }
  }

  // Normalize and store: O^T lane holds (d = dt*16 + hi*8 + r, query = lo)
  const float inv = 1.0f / runl;
  unsigned short* orow =
      attnout + (size_t)(b * 2048 + q0 + lo) * 1024 + h * 64 + hi * 8;
#pragma unroll
  for (int dt = 0; dt < 4; dt++) {
    FragU pk;
#pragma unroll
    for (int r = 0; r < 8; r++) pk.s[r] = f32_bf16(o[dt][r] * inv);
    *(u32x4*)(orow + dt * 16) = pk.q[0];  // 8 contiguous bf16, 16B aligned
  }
}

// ---------------------------------------------------------------------------
// Host launcher. Workspace layout (bytes):
//   Xb    @ 0         : 8192*1024*2  = 16,777,216
//   WqkvT @ 16777216  : 3072*1024*2  =  6,291,456
//   WoutT @ 23068672  : 1024*1024*2  =  2,097,152
//   QKVb  @ 25165824  : 8192*3072*2  = 50,331,648
//   Vt    @ 75497472  : 64*64*2048*2 = 16,777,216
//   Attn  @ 92274688  : 8192*1024*2  = 16,777,216   (total ~104 MB, L2-resident)
// ---------------------------------------------------------------------------
extern "C" void kernel_launch(void* const* d_in, const int* in_sizes, int n_in,
                              void* d_out, int out_size, void* d_ws,
                              size_t ws_size, hipStream_t stream) {
  const float* x     = (const float*)d_in[0];
  const float* w_qkv = (const float*)d_in[1];
  const float* w_out = (const float*)d_in[2];
  float* out = (float*)d_out;

  char* ws = (char*)d_ws;
  unsigned short* Xb    = (unsigned short*)(ws);
  unsigned short* WqkvT = (unsigned short*)(ws + (size_t)16777216);
  unsigned short* WoutT = (unsigned short*)(ws + (size_t)23068672);
  unsigned short* QKVb  = (unsigned short*)(ws + (size_t)25165824);
  unsigned short* Vt    = (unsigned short*)(ws + (size_t)75497472);
  unsigned short* Attn  = (unsigned short*)(ws + (size_t)92274688);

  // x -> bf16 (8 elems/thread)
  cvt_bf16_kernel<<<(8192 * 1024 / 8) / 256, 256, 0, stream>>>(x, Xb,
                                                               8192 * 1024 / 8);
  cvt_transpose_kernel<<<dim3(3072 / 32, 1024 / 32), 256, 0, stream>>>(
      w_qkv, WqkvT, 1024, 3072);
  cvt_transpose_kernel<<<dim3(1024 / 32, 1024 / 32), 256, 0, stream>>>(
      w_out, WoutT, 1024, 1024);

  // QKV = Xb @ WqkvT^T : (8192 x 3072), bf16 out
  gemm_wmma<true><<<dim3(64, 24), 128, 0, stream>>>(Xb, WqkvT, (void*)QKVb,
                                                    8192, 3072, 1024);

  build_vt_kernel<<<dim3(64, 32), 256, 0, stream>>>(QKVb, Vt);

  // grid: (B*H = 64, query-chunk groups = 128/4 = 32), 4 waves/block
  attn_wmma<<<dim3(64, 32), 128, 0, stream>>>(QKVb, Vt, Attn);

  // Out = Attn @ WoutT^T : (8192 x 1024), f32 out
  gemm_wmma<false><<<dim3(64, 8), 128, 0, stream>>>(Attn, WoutT, (void*)out,
                                                    8192, 1024, 1024);
}
